// HyenaY_1236950582229
// MI455X (gfx1250) — compile-verified, matrix operand order checked by reading, not run on hardware
//
#include <hip/hip_runtime.h>
#include <hip/hip_bf16.h>

// ---------------------------------------------------------------- constants
#define S_LEN 2048
#define D_DIM 2048
#define L_FILT 7
#define H_HEADS 16
#define HD_DIM 128
#define KVD 512          // KV heads (4) * HD (128)
#define FF_DIM 8192

typedef __bf16 bf16;
typedef __attribute__((ext_vector_type(16))) __bf16 bf16x16;
typedef __attribute__((ext_vector_type(8)))  __bf16 bf16x8;
typedef __attribute__((ext_vector_type(8)))  float  f32x8;
typedef __attribute__((ext_vector_type(4)))  unsigned int u32x4;
typedef __attribute__((ext_vector_type(8)))  int i32x8;
typedef __attribute__((ext_vector_type(4)))  int i32x4;

// -------- TDM availability (device pass only; host pass parses fallback)
#if defined(__has_builtin)
#if __has_builtin(__builtin_amdgcn_tensor_load_to_lds) && \
    __has_builtin(__builtin_amdgcn_s_wait_tensorcnt)
#define USE_TDM 1
#endif
#endif
#ifndef USE_TDM
#define USE_TDM 0
#endif

// ---------------------------------------------------------------- WMMA utils
static __device__ inline f32x8 wmma_bf16f32(bf16x16 a, bf16x16 b, f32x8 c) {
  // D(16x16 f32) = A(16x32 bf16) * B(32x16 bf16) + C
  return __builtin_amdgcn_wmma_f32_16x16x32_bf16(false, a, false, b,
                                                 (short)0, c, false, false);
}

// Load one 16x32 bf16 operand fragment from an LDS row (row-major).
// Per ISA 7.12.2: lane&15 selects the row; half = lane>>4 selects K slots
// {h*8..h*8+7} and {16+h*8..16+h*8+7}.  Two ds_load_b128.
static __device__ inline bf16x16 frag_ld(const bf16* __restrict__ row, int half) {
  bf16x8 lo = *(const bf16x8*)(row + half * 8);
  bf16x8 hi = *(const bf16x8*)(row + 16 + half * 8);
  bf16x16 f;
#pragma unroll
  for (int i = 0; i < 8; ++i) { f[i] = lo[i]; f[8 + i] = hi[i]; }
  return f;
}

// ---------------------------------------------------------------- TDM helper
#if USE_TDM
// 2D tile load Global -> LDS via Tensor Data Mover (ISA ch.8 D# layout).
// data_size = 2B.  LDS padding: every 16 DWORDs (64B = 32 bf16 = BK) insert
// 4 DWORDs (16B = 8 bf16) -> reproduces the LDT=40 padded row layout.
static __device__ inline void tdm_load_2d(unsigned lds_off, const void* gptr,
                                          unsigned tile_k, unsigned tile_rows,
                                          unsigned k_elems, unsigned rows,
                                          unsigned stride_elems) {
  unsigned long long ga = (unsigned long long)(uintptr_t)gptr;
  u32x4 g0 = {
      1u,                                         // count=1, user descriptor
      lds_off,                                    // lds_addr (bytes)
      (unsigned)(ga & 0xffffffffu),               // global_addr[31:0]
      (unsigned)((ga >> 32) & 0x01ffffffu) | (2u << 30)  // addr[56:32], type=2
  };
  i32x8 g1;
  g1[0] = (int)((1u << 16) |                      // data_size = 2B
                (1u << 20) |                      // pad_enable
                (3u << 22) |                      // pad_interval: 16 DWORDs
                (3u << 25));                      // pad_amount:   4 DWORDs
  g1[1] = (int)((k_elems & 0xffffu) << 16);       // tensor_dim0 lo16
  g1[2] = (int)(((k_elems >> 16) & 0xffffu) | ((rows & 0xffffu) << 16));
  g1[3] = (int)(((rows >> 16) & 0xffffu) | ((tile_k & 0xffffu) << 16));
  g1[4] = (int)(tile_rows & 0xffffu);             // tile_dim1 (tile_dim2=0)
  g1[5] = (int)stride_elems;                      // tensor_dim0_stride[31:0]
  g1[6] = 0;
  g1[7] = 0;
  i32x4 z4 = {0, 0, 0, 0};
#if __clang_major__ >= 23
  i32x8 z8 = {0, 0, 0, 0, 0, 0, 0, 0};
  __builtin_amdgcn_tensor_load_to_lds(g0, g1, z4, z4, z8, 0);
#else
  __builtin_amdgcn_tensor_load_to_lds(g0, g1, z4, z4, 0);
#endif
}
#endif  // USE_TDM

// ---------------------------------------------------------------- LayerNorm
__global__ __launch_bounds__(256)
void ln_kernel(const float* __restrict__ x, const float* __restrict__ g,
               const float* __restrict__ b, bf16* __restrict__ out) {
  int row = blockIdx.x;
  const float* xr = x + (size_t)row * D_DIM;
  float v[8];
  float s = 0.f, s2 = 0.f;
#pragma unroll
  for (int j = 0; j < 8; ++j) {
    v[j] = xr[threadIdx.x + j * 256];
    s += v[j];
    s2 += v[j] * v[j];
  }
#pragma unroll
  for (int off = 16; off >= 1; off >>= 1) {
    s  += __shfl_xor(s, off, 32);
    s2 += __shfl_xor(s2, off, 32);
  }
  __shared__ float rs[8], rs2[8];
  int wave = threadIdx.x >> 5, lane = threadIdx.x & 31;
  if (lane == 0) { rs[wave] = s; rs2[wave] = s2; }
  __syncthreads();
  s = 0.f; s2 = 0.f;
#pragma unroll
  for (int w = 0; w < 8; ++w) { s += rs[w]; s2 += rs2[w]; }
  float mean = s * (1.f / D_DIM);
  float var  = s2 * (1.f / D_DIM) - mean * mean;
  float rstd = rsqrtf(var + 1e-5f);
  bf16* orow = out + (size_t)row * D_DIM;
#pragma unroll
  for (int j = 0; j < 8; ++j) {
    int d = threadIdx.x + j * 256;
    orow[d] = (bf16)((v[j] - mean) * rstd * g[d] + b[d]);
  }
}

// ------------------------------------------- weight transpose+convert to bf16
// In: W fp32 [K x N] row-major.  Out: Wt bf16 [N x K] row-major.
__global__ __launch_bounds__(256)
void tconv_kernel(const float* __restrict__ W, bf16* __restrict__ Wt,
                  int K, int N) {
  __shared__ float t[32][33];
  int n0 = blockIdx.x * 32, k0 = blockIdx.y * 32;
  int tx = threadIdx.x, ty = threadIdx.y;  // 32 x 8
#pragma unroll
  for (int j = 0; j < 4; ++j)
    t[ty + j * 8][tx] = W[(size_t)(k0 + ty + j * 8) * N + n0 + tx];
  __syncthreads();
#pragma unroll
  for (int j = 0; j < 4; ++j)
    Wt[(size_t)(n0 + ty + j * 8) * K + k0 + tx] = (bf16)t[tx][ty + j * 8];
}

// ---------------------------------------------------------------- WMMA GEMM
// C[M x N] = act(A[M x K](bf16) * Wt[N x K]^T(bf16) + bias) (+ resid)
// 128x128 tile / workgroup, BK=32, 8 waves in 4(m) x 2(n) grid.
// Tile staging: TDM tensor_load_to_lds double-buffered (wave 0 issues tile
// k+1 while all waves run WMMAs on tile k); vector-load fallback otherwise.
#define BM 128
#define BN 128
#define BK 32
#define LDT 40  // BK + 8 bf16 pad (matches TDM pad_interval/pad_amount)

template <int ACT, int RES, int OBF>
__global__ __launch_bounds__(256)
void gemm_kernel(const bf16* __restrict__ A, const bf16* __restrict__ Wt,
                 const float* __restrict__ bias, const float* __restrict__ resid,
                 float* __restrict__ Cf, bf16* __restrict__ Cb,
                 int M, int N, int K) {
  int m0 = blockIdx.y * BM, n0 = blockIdx.x * BN;
  int t = threadIdx.x;
  int wave = t >> 5, lane = t & 31;
  int wm = wave & 3, wn = wave >> 2;
  int r = lane & 15, h = lane >> 4;

  f32x8 acc[2][4];
#pragma unroll
  for (int mi = 0; mi < 2; ++mi)
#pragma unroll
    for (int ni = 0; ni < 4; ++ni)
#pragma unroll
      for (int i = 0; i < 8; ++i) acc[mi][ni][i] = 0.f;

#if USE_TDM
  __shared__ __align__(16) bf16 As[2][BM * LDT];
  __shared__ __align__(16) bf16 Bs[2][BN * LDT];
  const bf16* Arow = A  + (size_t)m0 * K;
  const bf16* Brow = Wt + (size_t)n0 * K;
  bool issuer = (wave == 0);
  if (issuer) {
    tdm_load_2d((unsigned)(uintptr_t)&As[0][0], Arow, BK, BM, K, M, K);
    tdm_load_2d((unsigned)(uintptr_t)&Bs[0][0], Brow, BK, BN, K, N, K);
  }
  int cur = 0;
  for (int k0 = 0; k0 < K; k0 += BK) {
    if (issuer) {
      if (k0 + BK < K) {
        tdm_load_2d((unsigned)(uintptr_t)&As[cur ^ 1][0], Arow + k0 + BK,
                    BK, BM, K, M, K);
        tdm_load_2d((unsigned)(uintptr_t)&Bs[cur ^ 1][0], Brow + k0 + BK,
                    BK, BN, K, N, K);
        __builtin_amdgcn_s_wait_tensorcnt((short)2);  // current tile done
      } else {
        __builtin_amdgcn_s_wait_tensorcnt((short)0);
      }
    }
    __syncthreads();
    bf16x16 af[2], bfv[4];
#pragma unroll
    for (int mi = 0; mi < 2; ++mi)
      af[mi] = frag_ld(&As[cur][(wm * 32 + mi * 16 + r) * LDT], h);
#pragma unroll
    for (int ni = 0; ni < 4; ++ni)
      bfv[ni] = frag_ld(&Bs[cur][(wn * 64 + ni * 16 + r) * LDT], h);
#pragma unroll
    for (int mi = 0; mi < 2; ++mi)
#pragma unroll
      for (int ni = 0; ni < 4; ++ni)
        acc[mi][ni] = wmma_bf16f32(af[mi], bfv[ni], acc[mi][ni]);
    __syncthreads();
    cur ^= 1;
  }
#else
  __shared__ __align__(16) bf16 As[BM * LDT];
  __shared__ __align__(16) bf16 Bs[BN * LDT];
  int srow = t >> 1, scol = (t & 1) * 16;
  const bf16* Ag = A  + (size_t)(m0 + srow) * K + scol;
  const bf16* Bg = Wt + (size_t)(n0 + srow) * K + scol;
  bf16* AsP = &As[srow * LDT + scol];
  bf16* BsP = &Bs[srow * LDT + scol];
  for (int k0 = 0; k0 < K; k0 += BK) {
    *(uint4*)AsP       = *(const uint4*)Ag;
    *(uint4*)(AsP + 8) = *(const uint4*)(Ag + 8);
    *(uint4*)BsP       = *(const uint4*)Bg;
    *(uint4*)(BsP + 8) = *(const uint4*)(Bg + 8);
    Ag += BK; Bg += BK;
    if (k0 + BK < K) {
      __builtin_prefetch(Ag, 0, 1);
      __builtin_prefetch(Bg, 0, 1);
    }
    __syncthreads();
    bf16x16 af[2], bfv[4];
#pragma unroll
    for (int mi = 0; mi < 2; ++mi)
      af[mi] = frag_ld(&As[(wm * 32 + mi * 16 + r) * LDT], h);
#pragma unroll
    for (int ni = 0; ni < 4; ++ni)
      bfv[ni] = frag_ld(&Bs[(wn * 64 + ni * 16 + r) * LDT], h);
#pragma unroll
    for (int mi = 0; mi < 2; ++mi)
#pragma unroll
      for (int ni = 0; ni < 4; ++ni)
        acc[mi][ni] = wmma_bf16f32(af[mi], bfv[ni], acc[mi][ni]);
    __syncthreads();
  }
#endif

  // Epilogue.  C layout: lane&15 = N col, vgpr i + 8*(lane>>4) = M row.
#pragma unroll
  for (int mi = 0; mi < 2; ++mi) {
#pragma unroll
    for (int ni = 0; ni < 4; ++ni) {
      int n = n0 + wn * 64 + ni * 16 + r;
      float bia = bias ? bias[n] : 0.f;
#pragma unroll
      for (int i = 0; i < 8; ++i) {
        int m = m0 + wm * 32 + mi * 16 + h * 8 + i;
        float v = acc[mi][ni][i] + bia;
        if (ACT == 1) v = v / (1.f + __expf(-v));   // SiLU
        if (RES) v += resid[(size_t)m * N + n];
        if (OBF) Cb[(size_t)m * N + n] = (bf16)v;
        else     Cf[(size_t)m * N + n] = v;
      }
    }
  }
}

// ------------------------------------- causal depthwise conv + gate + resid
// out[s,d] = x[s,d] + v[s,d] * sum_l filt[d,l] * u[s+l-6,d]
__global__ __launch_bounds__(256)
void conv_gate_kernel(const float* __restrict__ x, const float* __restrict__ u,
                      const float* __restrict__ v, const float* __restrict__ filt,
                      float* __restrict__ out) {
  int idx = blockIdx.x * 256 + threadIdx.x;
  int s = idx >> 11;        // / D_DIM
  int d = idx & (D_DIM - 1);
  float acc = 0.f;
#pragma unroll
  for (int l = 0; l < L_FILT; ++l) {
    int sp = s + l - (L_FILT - 1);
    if (sp >= 0) acc += filt[d * L_FILT + l] * u[(size_t)sp * D_DIM + d];
  }
  out[idx] = x[idx] + acc * v[idx];
}

// ---------------------------------------------------------------- Flash GQA
// Grid: (S/128, H).  8 waves, each owns 16 query rows.
// S^T = K * Q^T so softmax key-reduction is in-lane + one xor16 shuffle;
// exp'd P repacks into the O += P*V A-fragment register-only.
// Wave-uniform branches skip masking on interior blocks and skip compute on
// fully-masked blocks (EXEC stays all-ones around WMMAs).
__global__ __launch_bounds__(256)
void flash_kernel(const bf16* __restrict__ Q, const bf16* __restrict__ Kg,
                  const bf16* __restrict__ Vg, bf16* __restrict__ O) {
  __shared__ __align__(16) bf16 Ks[32 * 136];   // [32 keys][128 hd + pad]
  __shared__ __align__(16) bf16 Vs[128 * 40];   // [128 d][32 keys + pad]
  int q0 = blockIdx.x * 128;
  int hq = blockIdx.y;
  int kvh = hq >> 2;                // G = 4 query heads per kv head
  int t = threadIdx.x, wave = t >> 5, lane = t & 31;
  int r = lane & 15, h = lane >> 4;
  int qlo = q0 + wave * 16;         // wave's lowest query row
  int qi = qlo + r;                 // this lane's query row (both halves)
  const float scale = 0.08838834764831845f;  // HD^-0.5

  // Q^T B-fragments, 4 chunks of 32 along HD, scale folded into the load.
  bf16x16 qf[4];
  {
    const bf16* qrow = Q + (size_t)qi * D_DIM + hq * HD_DIM;
#pragma unroll
    for (int c = 0; c < 4; ++c) {
      bf16x8 lo = *(const bf16x8*)(qrow + c * 32 + h * 8);
      bf16x8 hi = *(const bf16x8*)(qrow + c * 32 + 16 + h * 8);
#pragma unroll
      for (int i = 0; i < 8; ++i) {
        qf[c][i]     = (bf16)((float)lo[i] * scale);
        qf[c][8 + i] = (bf16)((float)hi[i] * scale);
      }
    }
  }

  f32x8 o_acc[8];                   // O tile: lane = d col, vgpr = q row
#pragma unroll
  for (int ni = 0; ni < 8; ++ni)
#pragma unroll
    for (int i = 0; i < 8; ++i) o_acc[ni][i] = 0.f;
  float m = -1e30f, lsum = 0.f;

  int kend = q0 + 128;              // causal upper bound for this block
  for (int kb = 0; kb < kend; kb += 32) {
    {  // cooperative stage: K tile row-major, V tile transposed
      int row = t >> 3, col = (t & 7) * 16;
      const bf16* kg = Kg + (size_t)(kb + row) * KVD + kvh * HD_DIM + col;
      *(uint4*)&Ks[row * 136 + col]     = *(const uint4*)kg;
      *(uint4*)&Ks[row * 136 + col + 8] = *(const uint4*)(kg + 8);
      const bf16* vg = Vg + (size_t)(kb + row) * KVD + kvh * HD_DIM + col;
      bf16x8 v0 = *(const bf16x8*)vg;
      bf16x8 v1 = *(const bf16x8*)(vg + 8);
#pragma unroll
      for (int j = 0; j < 8; ++j) {
        Vs[(col + j) * 40 + row]     = v0[j];
        Vs[(col + 8 + j) * 40 + row] = v1[j];
      }
    }
    __syncthreads();

    if (kb <= qlo + 15) {           // wave-uniform: block not fully masked
      // S^T tiles: keys kb..kb+15 (st0), kb+16..kb+31 (st1); lane = query.
      f32x8 st0, st1;
#pragma unroll
      for (int i = 0; i < 8; ++i) { st0[i] = 0.f; st1[i] = 0.f; }
#pragma unroll
      for (int c = 0; c < 4; ++c) {
        bf16x16 k0f = frag_ld(&Ks[r * 136 + c * 32], h);
        bf16x16 k1f = frag_ld(&Ks[(16 + r) * 136 + c * 32], h);
        st0 = wmma_bf16f32(k0f, qf[c], st0);
        st1 = wmma_bf16f32(k1f, qf[c], st1);
      }

      float mt = -1e30f;
      if (kb + 31 > qlo) {          // wave-uniform: diagonal block -> mask
#pragma unroll
        for (int i = 0; i < 8; ++i) {
          int key0 = kb + h * 8 + i;
          if (key0 > qi)      st0[i] = -1e30f;
          if (key0 + 16 > qi) st1[i] = -1e30f;
          mt = fmaxf(mt, fmaxf(st0[i], st1[i]));
        }
      } else {                      // interior block: no mask VALU
#pragma unroll
        for (int i = 0; i < 8; ++i) mt = fmaxf(mt, fmaxf(st0[i], st1[i]));
      }
      mt = fmaxf(mt, __shfl_xor(mt, 16, 32));
      float mn = fmaxf(m, mt);
      float alpha = __expf(m - mn);
      float p0[8], p1[8];
      float rsum = 0.f;
#pragma unroll
      for (int i = 0; i < 8; ++i) {
        p0[i] = __expf(st0[i] - mn);
        p1[i] = __expf(st1[i] - mn);
        rsum += p0[i] + p1[i];
      }
      rsum += __shfl_xor(rsum, 16, 32);
      lsum = lsum * alpha + rsum;
      m = mn;

      // Broadcast alpha per O-accumulator row (q = vgpr + 8*half).
      float ab[8];
#pragma unroll
      for (int i = 0; i < 8; ++i) ab[i] = __shfl(alpha, i + (h << 3), 32);
#pragma unroll
      for (int ni = 0; ni < 8; ++ni)
#pragma unroll
        for (int i = 0; i < 8; ++i) o_acc[ni][i] *= ab[i];

      // Register-only repack of P into the A-fragment for O += P * V.
      bf16x16 pf;
#pragma unroll
      for (int i = 0; i < 8; ++i) {
        pf[i]     = (bf16)p0[i];
        pf[8 + i] = (bf16)p1[i];
      }
#pragma unroll
      for (int ni = 0; ni < 8; ++ni) {
        bf16x16 vf = frag_ld(&Vs[(ni * 16 + r) * 40], h);
        o_acc[ni] = wmma_bf16f32(pf, vf, o_acc[ni]);
      }
    }
    __syncthreads();
  }

  float lb[8];
#pragma unroll
  for (int i = 0; i < 8; ++i) lb[i] = __shfl(lsum, i + (h << 3), 32);
#pragma unroll
  for (int ni = 0; ni < 8; ++ni) {
#pragma unroll
    for (int i = 0; i < 8; ++i) {
      int q = q0 + wave * 16 + h * 8 + i;
      int dcol = hq * HD_DIM + ni * 16 + r;
      O[(size_t)q * D_DIM + dcol] = (bf16)(o_acc[ni][i] / lb[i]);
    }
  }
}

// ---------------------------------------------------------------- launcher
extern "C" void kernel_launch(void* const* d_in, const int* in_sizes, int n_in,
                              void* d_out, int out_size, void* d_ws,
                              size_t ws_size, hipStream_t stream) {
  (void)in_sizes; (void)n_in; (void)out_size; (void)ws_size;
  const int S = S_LEN, D = D_DIM, FF = FF_DIM;

  // Inputs in setup_inputs() insertion order: x, hyena[0](7), hyena[1](7), gqa(16)
  const float* x = (const float*)d_in[0];
  struct HyP { const float *g1, *b1, *Wu, *bu, *Wv, *bv, *filt; } hy[2];
  for (int i = 0; i < 2; ++i) {
    int o = 1 + i * 7;
    hy[i].g1 = (const float*)d_in[o + 0];
    hy[i].b1 = (const float*)d_in[o + 1];
    hy[i].Wu = (const float*)d_in[o + 2];
    hy[i].bu = (const float*)d_in[o + 3];
    hy[i].Wv = (const float*)d_in[o + 4];
    hy[i].bv = (const float*)d_in[o + 5];
    hy[i].filt = (const float*)d_in[o + 6];
  }
  const float* gg1 = (const float*)d_in[15];
  const float* gb1 = (const float*)d_in[16];
  const float* Wq  = (const float*)d_in[17];
  const float* bq  = (const float*)d_in[18];
  const float* Wk  = (const float*)d_in[19];
  const float* bk  = (const float*)d_in[20];
  const float* Wv  = (const float*)d_in[21];
  const float* bv  = (const float*)d_in[22];
  const float* Wo  = (const float*)d_in[23];
  const float* bo  = (const float*)d_in[24];
  const float* gg2 = (const float*)d_in[25];
  const float* gb2 = (const float*)d_in[26];
  const float* W1  = (const float*)d_in[27];
  const float* b1f = (const float*)d_in[28];
  const float* W2  = (const float*)d_in[29];
  const float* b2f = (const float*)d_in[30];

  // Workspace bump allocator (d_ws is 256B aligned from hipMalloc).
  char* p = (char*)d_ws;
  auto alloc = [&](size_t bytes) -> void* {
    void* r = (void*)p;
    p += (bytes + 255) & ~(size_t)255;
    return r;
  };
  bf16*  hb   = (bf16*)alloc((size_t)S * D * 2);
  float* u    = (float*)alloc((size_t)S * D * 4);
  float* vbuf = (float*)alloc((size_t)S * D * 4);
  float* xa   = (float*)alloc((size_t)S * D * 4);
  float* xb   = (float*)alloc((size_t)S * D * 4);
  bf16*  qb   = (bf16*)alloc((size_t)S * D * 2);
  bf16*  kbb  = (bf16*)alloc((size_t)S * KVD * 2);
  bf16*  vbb  = (bf16*)alloc((size_t)S * KVD * 2);
  bf16*  ob   = (bf16*)alloc((size_t)S * D * 2);
  bf16*  f1   = (bf16*)alloc((size_t)S * FF * 2);
  bf16*  wt   = (bf16*)alloc((size_t)D * FF * 2);  // reused for every weight

  auto tc = [&](const float* W, int K, int N) {
    tconv_kernel<<<dim3(N / 32, K / 32), dim3(32, 8), 0, stream>>>(W, wt, K, N);
  };
  auto ggrid = [](int M, int N) { return dim3(N / 128, M / 128); };

  // ---- Hyena block 0: x -> xa
  ln_kernel<<<S, 256, 0, stream>>>(x, hy[0].g1, hy[0].b1, hb);
  tc(hy[0].Wu, D, D);
  gemm_kernel<0, 0, 0><<<ggrid(S, D), 256, 0, stream>>>(hb, wt, hy[0].bu, nullptr, u, nullptr, S, D, D);
  tc(hy[0].Wv, D, D);
  gemm_kernel<0, 0, 0><<<ggrid(S, D), 256, 0, stream>>>(hb, wt, hy[0].bv, nullptr, vbuf, nullptr, S, D, D);
  conv_gate_kernel<<<(S * D) / 256, 256, 0, stream>>>(x, u, vbuf, hy[0].filt, xa);

  // ---- Hyena block 1: xa -> xb
  ln_kernel<<<S, 256, 0, stream>>>(xa, hy[1].g1, hy[1].b1, hb);
  tc(hy[1].Wu, D, D);
  gemm_kernel<0, 0, 0><<<ggrid(S, D), 256, 0, stream>>>(hb, wt, hy[1].bu, nullptr, u, nullptr, S, D, D);
  tc(hy[1].Wv, D, D);
  gemm_kernel<0, 0, 0><<<ggrid(S, D), 256, 0, stream>>>(hb, wt, hy[1].bv, nullptr, vbuf, nullptr, S, D, D);
  conv_gate_kernel<<<(S * D) / 256, 256, 0, stream>>>(xa, u, vbuf, hy[1].filt, xb);

  // ---- GQA block: xb -> d_out
  ln_kernel<<<S, 256, 0, stream>>>(xb, gg1, gb1, hb);
  tc(Wq, D, D);
  gemm_kernel<0, 0, 1><<<ggrid(S, D), 256, 0, stream>>>(hb, wt, bq, nullptr, nullptr, qb, S, D, D);
  tc(Wk, D, KVD);
  gemm_kernel<0, 0, 1><<<ggrid(S, KVD), 256, 0, stream>>>(hb, wt, bk, nullptr, nullptr, kbb, S, KVD, D);
  tc(Wv, D, KVD);
  gemm_kernel<0, 0, 1><<<ggrid(S, KVD), 256, 0, stream>>>(hb, wt, bv, nullptr, nullptr, vbb, S, KVD, D);
  flash_kernel<<<dim3(S / 128, H_HEADS), 256, 0, stream>>>(qb, kbb, vbb, ob);
  tc(Wo, D, D);
  gemm_kernel<0, 1, 0><<<ggrid(S, D), 256, 0, stream>>>(ob, wt, bo, xb, xa, nullptr, S, D, D);
  ln_kernel<<<S, 256, 0, stream>>>(xa, gg2, gb2, hb);
  tc(W1, D, FF);
  gemm_kernel<1, 0, 1><<<ggrid(S, FF), 256, 0, stream>>>(hb, wt, b1f, nullptr, nullptr, f1, S, FF, D);
  tc(W2, FF, D);
  gemm_kernel<0, 1, 0><<<ggrid(S, D), 256, 0, stream>>>(f1, wt, b2f, xa, (float*)d_out, nullptr, S, D, FF);
}
